// Sinkhorn_Decoder5_34832184770742
// MI455X (gfx1250) — compile-verified, matrix operand order checked by reading, not run on hardware
//
#include <hip/hip_runtime.h>
#include <hip/hip_bf16.h>

typedef __attribute__((ext_vector_type(16))) _Float16 v16h;
typedef __attribute__((ext_vector_type(8)))  _Float16 h8;
typedef __attribute__((ext_vector_type(8)))  float    v8f;

#define MAXN 9
#define NPAIR 36
#define NLOC 45
#define BATCH 4096
#define NROWS (BATCH * NLOC)   // 184320

__constant__ int PI_c[NPAIR] = {0,0,0,0,0,0,0,0, 1,1,1,1,1,1,1, 2,2,2,2,2,2,
                                3,3,3,3,3, 4,4,4,4, 5,5,5, 6,6, 7};
__constant__ int PJ_c[NPAIR] = {1,2,3,4,5,6,7,8, 2,3,4,5,6,7,8, 3,4,5,6,7,8,
                                4,5,6,7,8, 5,6,7,8, 6,7,8, 7,8, 8};

enum { EPI_NONE, EPI_BIAS, EPI_BIAS_RELU, EPI_BIAS_LRELU, EPI_BIAS_TANH,
       EPI_BIAS_BN_LRELU, EPI_LRELU };

// ---- weight prep: f32 [K,fo] row-major  ->  f16 transposed+padded [Npad,Kpad]
__global__ void prep_wt(const float* __restrict__ W, _Float16* __restrict__ Wt,
                        int K, int fo, int Kpad, int Npad) {
    int i = blockIdx.x * blockDim.x + threadIdx.x;
    if (i >= Npad * Kpad) return;
    int n = i / Kpad, k = i % Kpad;
    float v = (n < fo && k < K) ? W[(size_t)k * fo + n] : 0.0f;
    Wt[i] = (_Float16)v;
}

__global__ void f32_to_f16(const float* __restrict__ in, _Float16* __restrict__ out, int n) {
    int i = blockIdx.x * blockDim.x + threadIdx.x;
    if (i < n) out[i] = (_Float16)in[i];
}

// ---- WMMA GEMM: each wave computes a 16 x (NT*16) strip, A fragment reused NT times
template<int EPI, bool F32OUT, int NT>
__global__ __launch_bounds__(256) void gemm16(
    const _Float16* __restrict__ A, int Kpad,
    const _Float16* __restrict__ Bt,          // [Npad, Kpad] (transposed weights)
    const float* __restrict__ bias,
    const float* __restrict__ gamma,
    const float* __restrict__ beta,
    void* __restrict__ Cv, int ldc, int Mtiles, int Ngroups, int fo)
{
    const int wave = threadIdx.x >> 5;
    const int lane = threadIdx.x & 31;
    int tile = blockIdx.x * 8 + wave;
    if (tile >= Mtiles * Ngroups) return;          // wave-uniform: EXEC stays all-1s
    const int gidx = tile / Mtiles;
    const int mt   = tile - gidx * Mtiles;
    const int nt0  = gidx * NT;
    const int l16  = lane & 15;
    const int half = lane >> 4;

    const _Float16* Arow = A  + (size_t)(mt * 16 + l16) * Kpad;
    const _Float16* Brow = Bt + (size_t)(nt0 * 16 + l16) * Kpad;
    const size_t bstride = (size_t)16 * Kpad;      // one N-tile of weights

    v8f acc[NT] = {};
    for (int k0 = 0; k0 < Kpad; k0 += 32) {
        // CDNA5 16-bit A fragment: two contiguous 8-half runs per lane
        h8 alo = *(const h8*)(Arow + k0 + (half << 3));
        h8 ahi = *(const h8*)(Arow + k0 + 16 + (half << 3));
        v16h a = __builtin_shufflevector(alo, ahi, 0,1,2,3,4,5,6,7,8,9,10,11,12,13,14,15);
#pragma unroll
        for (int j = 0; j < NT; ++j) {
            const _Float16* Bj = Brow + (size_t)j * bstride;
            h8 blo = *(const h8*)(Bj + k0 + (half << 3));
            h8 bhi = *(const h8*)(Bj + k0 + 16 + (half << 3));
            v16h b = __builtin_shufflevector(blo, bhi, 0,1,2,3,4,5,6,7,8,9,10,11,12,13,14,15);
            acc[j] = __builtin_amdgcn_wmma_f32_16x16x32_f16(false, a, false, b,
                                                            (short)0, acc[j], false, false);
        }
    }

#pragma unroll
    for (int j = 0; j < NT; ++j) {
        const int cn = (nt0 + j) * 16 + l16;
        float bb = 0.0f, gg = 1.0f, be = 0.0f;
        if (cn < fo) {
            if (EPI == EPI_BIAS || EPI == EPI_BIAS_RELU || EPI == EPI_BIAS_LRELU ||
                EPI == EPI_BIAS_TANH || EPI == EPI_BIAS_BN_LRELU) bb = bias[cn];
            if (EPI == EPI_BIAS_BN_LRELU) { gg = gamma[cn]; be = beta[cn]; }
        }
#pragma unroll
        for (int r = 0; r < 8; ++r) {
            float v = acc[j][r];
            if (EPI == EPI_BIAS || EPI == EPI_BIAS_RELU || EPI == EPI_BIAS_LRELU ||
                EPI == EPI_BIAS_TANH || EPI == EPI_BIAS_BN_LRELU) v += bb;
            if (EPI == EPI_BIAS_RELU)  v = v > 0.0f ? v : 0.0f;
            if (EPI == EPI_BIAS_TANH)  v = tanhf(v);
            if (EPI == EPI_BIAS_BN_LRELU) { v = gg * v + be; v = v >= 0.0f ? v : 0.01f * v; }
            if (EPI == EPI_BIAS_LRELU || EPI == EPI_LRELU) v = v >= 0.0f ? v : 0.01f * v;
            int row = mt * 16 + half * 8 + r;
            if (cn < fo) {
                if (F32OUT) ((float*)Cv)[(size_t)row * ldc + cn] = v;
                else        ((_Float16*)Cv)[(size_t)row * ldc + cn] = (_Float16)v;
            }
        }
    }
}

// ---- build x0[N,96] from emb[B,576]: 9 node rows + 36 averaged edge rows, zero-pad 64..95
__global__ void build_x0(const _Float16* __restrict__ emb, _Float16* __restrict__ x) {
    int idx = blockIdx.x * blockDim.x + threadIdx.x;
    const int total = NROWS * 96;
    if (idx >= total) return;
    int f = idx % 96;
    int node = idx / 96;
    int g = node / NLOC, loc = node % NLOC;
    float v = 0.0f;
    if (f < 64) {
        const _Float16* eg = emb + (size_t)g * 576;
        if (loc < MAXN) {
            v = (float)eg[loc * 64 + f];
        } else {
            int p = loc - MAXN;
            v = 0.5f * ((float)eg[PI_c[p] * 64 + f] + (float)eg[PJ_c[p] * 64 + f]);
        }
    }
    x[idx] = (_Float16)v;
}

// ---- GIN aggregation: hh = (1+eps)*x + segment_sum over fixed 45-node topology
__global__ __launch_bounds__(256) void gin_agg(const _Float16* __restrict__ x, int ld,
                                               const float* __restrict__ eps_p,
                                               _Float16* __restrict__ hh) {
    __shared__ _Float16 sx[NLOC * 128];
    int g = blockIdx.x;
    const _Float16* xg = x + (size_t)g * NLOC * ld;
    for (int i = threadIdx.x; i < NLOC * ld; i += blockDim.x) sx[i] = xg[i];
    __syncthreads();
    float eps1 = 1.0f + eps_p[0];
    _Float16* hg = hh + (size_t)g * NLOC * ld;
    for (int i = threadIdx.x; i < NLOC * ld; i += blockDim.x) {
        int loc = i / ld, f = i % ld;
        float agg = 0.0f;
        if (loc < MAXN) {
            // node loc receives from every edge-node containing it (8 of them)
            for (int j = 0; j < MAXN; ++j) {
                if (j == loc) continue;
                int a = loc < j ? loc : j;
                int b = loc < j ? j : loc;
                int p = a * (2 * MAXN - a - 1) / 2 + (b - a - 1);
                agg += (float)sx[(MAXN + p) * ld + f];
            }
        } else {
            int p = loc - MAXN;
            agg = (float)sx[PI_c[p] * ld + f] + (float)sx[PJ_c[p] * ld + f];
        }
        hg[i] = (_Float16)(eps1 * (float)sx[i] + agg);
    }
}

// ---- node-count head tail: [B,64] @ [64,1] + b
__global__ void nnl_last(const _Float16* __restrict__ h, const float* __restrict__ W,
                         const float* __restrict__ b, float* __restrict__ out) {
    int r = blockIdx.x * blockDim.x + threadIdx.x;
    if (r >= BATCH) return;
    float acc = b[0];
    for (int k = 0; k < 64; ++k) acc += (float)h[(size_t)r * 64 + k] * W[k];
    out[r] = acc;
}

// ---- scatter out10[N,16] -> x_nodes [B*9,5] ++ edge_attr [B*36,5]
__global__ void scatter_out(const float* __restrict__ o, float* __restrict__ out) {
    int idx = blockIdx.x * blockDim.x + threadIdx.x;
    const int n_node = BATCH * MAXN * 5;     // 184320
    const int n_edge = BATCH * NPAIR * 5;    // 737280
    if (idx >= n_node + n_edge) return;
    if (idx < n_node) {
        int c = idx % 5, nn = idx / 5;
        int g = nn / MAXN, i = nn % MAXN;
        out[idx] = o[((size_t)g * NLOC + i) * 16 + c];
    } else {
        int e = idx - n_node;
        int c = e % 5, ee = e / 5;
        int g = ee / NPAIR, p = ee % NPAIR;
        out[idx] = o[((size_t)g * NLOC + MAXN + p) * 16 + 5 + c];
    }
}

// ----------------------------------------------------------------------------
template<int EPI, bool F32, int NT>
static inline void gemm_launch(hipStream_t s, const void* A, const void* Bt, int Kpad,
                               const float* bias, const float* gamma, const float* beta,
                               void* C, int M, int fo, int Npad) {
    int Mtiles = M / 16, Ngroups = (Npad / 16) / NT;
    int total = Mtiles * Ngroups;
    gemm16<EPI, F32, NT><<<dim3((total + 7) / 8), dim3(256), 0, s>>>(
        (const _Float16*)A, Kpad, (const _Float16*)Bt, bias, gamma, beta,
        C, Npad, Mtiles, Ngroups, fo);
}

extern "C" void kernel_launch(void* const* d_in, const int* in_sizes, int n_in,
                              void* d_out, int out_size, void* d_ws, size_t ws_size,
                              hipStream_t stream) {
    (void)in_sizes; (void)n_in; (void)out_size;
    char* ws = (char*)d_ws;

    // ---- workspace layout (256B aligned)
    size_t off = 0;
    auto alloc = [&](size_t bytes) { size_t o = off; off += (bytes + 255) & ~(size_t)255; return o; };
    const size_t o_lat16 = alloc((size_t)BATCH * 128 * 2);
    const size_t o_x     = alloc((size_t)NROWS * 128 * 2);
    const size_t o_hh    = alloc((size_t)NROWS * 128 * 2);
    const size_t o_t     = alloc((size_t)NROWS * 128 * 2);
    const size_t o_big   = alloc((size_t)NROWS * 256 * 2);

    // weights: transposed f16, padded
    struct WDesc { int idx, K, fo, Kpad, Npad; };
    const WDesc wd[25] = {
        {2,128,128,128,128}, {4,128,256,128,256}, {6,256,512,256,512}, {8,512,576,512,576},
        {10,128,128,128,128}, {12,128,64,128,64},
        {17,69,32,96,32},   {19,32,32,32,32},
        {24,32,32,32,32},   {26,32,32,32,32},
        {31,32,64,32,64},   {33,64,64,64,64},
        {38,64,64,64,64},   {40,64,64,64,64},
        {45,64,128,64,128}, {47,128,128,128,128},
        {52,128,128,128,128},{54,128,128,128,128},
        {59,128,128,128,128},{61,128,128,128,128},
        {66,128,128,128,128},{68,128,128,128,128},
        {72,128,256,128,256},{73,256,128,256,128},{74,128,10,128,16},
    };
    size_t wt_off[25];
    for (int i = 0; i < 25; ++i)
        wt_off[i] = alloc((size_t)wd[i].Kpad * wd[i].Npad * 2);
    if (ws_size < off) return;  // workspace too small: bail deterministically

    for (int i = 0; i < 25; ++i) {
        int total = wd[i].Kpad * wd[i].Npad;
        prep_wt<<<dim3((total + 255) / 256), dim3(256), 0, stream>>>(
            (const float*)d_in[wd[i].idx], (_Float16*)(ws + wt_off[i]),
            wd[i].K, wd[i].fo, wd[i].Kpad, wd[i].Npad);
    }

    _Float16* lat16 = (_Float16*)(ws + o_lat16);
    _Float16* xbuf  = (_Float16*)(ws + o_x);
    _Float16* hhbuf = (_Float16*)(ws + o_hh);
    _Float16* tbuf  = (_Float16*)(ws + o_t);
    char* big = ws + o_big;
    _Float16* g0  = (_Float16*)big;
    _Float16* g1  = g0 + (size_t)BATCH * 128;
    _Float16* g2  = g1 + (size_t)BATCH * 256;
    _Float16* emb = g2 + (size_t)BATCH * 512;
    _Float16* nh1 = emb + (size_t)BATCH * 576;
    _Float16* nh2 = nh1 + (size_t)BATCH * 128;

    // latent f32 -> f16
    {
        int n = BATCH * 128;
        f32_to_f16<<<dim3((n + 255) / 256), dim3(256), 0, stream>>>((const float*)d_in[0], lat16, n);
    }
    float* outp = (float*)d_out;

    // ---- node-count head: 128->128 lrelu ->64 lrelu ->1
    gemm_launch<EPI_BIAS_LRELU,false,4>(stream, lat16, ws + wt_off[4], 128,
        (const float*)d_in[11], nullptr, nullptr, nh1, BATCH, 128, 128);
    gemm_launch<EPI_BIAS_LRELU,false,4>(stream, nh1, ws + wt_off[5], 128,
        (const float*)d_in[13], nullptr, nullptr, nh2, BATCH, 64, 64);
    nnl_last<<<dim3((BATCH + 255) / 256), dim3(256), 0, stream>>>(
        nh2, (const float*)d_in[14], (const float*)d_in[15],
        outp + (size_t)BATCH * MAXN * 5 + (size_t)BATCH * NPAIR * 5);

    // ---- generator: tanh 128->128->256->512, nodes 512->576
    gemm_launch<EPI_BIAS_TANH,false,4>(stream, lat16, ws + wt_off[0], 128,
        (const float*)d_in[3], nullptr, nullptr, g0, BATCH, 128, 128);
    gemm_launch<EPI_BIAS_TANH,false,4>(stream, g0, ws + wt_off[1], 128,
        (const float*)d_in[5], nullptr, nullptr, g1, BATCH, 256, 256);
    gemm_launch<EPI_BIAS_TANH,false,4>(stream, g1, ws + wt_off[2], 256,
        (const float*)d_in[7], nullptr, nullptr, g2, BATCH, 512, 512);
    gemm_launch<EPI_BIAS,false,4>(stream, g2, ws + wt_off[3], 512,
        (const float*)d_in[9], nullptr, nullptr, emb, BATCH, 576, 576);

    // ---- x0 [N,96]
    {
        int n = NROWS * 96;
        build_x0<<<dim3((n + 255) / 256), dim3(256), 0, stream>>>(emb, xbuf);
    }

    // ---- 8 GIN layers
    const int fipad[8] = {96, 32, 32, 64, 64, 128, 128, 128};
    const int fo_[8]   = {32, 32, 64, 64, 128, 128, 128, 128};
    for (int l = 0; l < 8; ++l) {
        int base = 16 + 7 * l;
        const float* eps   = (const float*)d_in[base + 0];
        const float* b1    = (const float*)d_in[base + 2];
        const float* b2    = (const float*)d_in[base + 4];
        const float* gam   = (const float*)d_in[base + 5];
        const float* bet   = (const float*)d_in[base + 6];
        void* W1t = ws + wt_off[6 + 2 * l];
        void* W2t = ws + wt_off[6 + 2 * l + 1];
        int ld = fipad[l], fo = fo_[l];

        gin_agg<<<dim3(BATCH), dim3(256), 0, stream>>>(xbuf, ld, eps, hhbuf);
        if (fo == 32) {
            gemm_launch<EPI_BIAS_RELU,false,2>(stream, hhbuf, W1t, ld,
                b1, nullptr, nullptr, tbuf, NROWS, fo, fo);
            if (l < 7)
                gemm_launch<EPI_BIAS_BN_LRELU,false,2>(stream, tbuf, W2t, fo,
                    b2, gam, bet, xbuf, NROWS, fo, fo);
            else
                gemm_launch<EPI_BIAS,false,2>(stream, tbuf, W2t, fo,
                    b2, nullptr, nullptr, xbuf, NROWS, fo, fo);
        } else {
            gemm_launch<EPI_BIAS_RELU,false,4>(stream, hhbuf, W1t, ld,
                b1, nullptr, nullptr, tbuf, NROWS, fo, fo);
            if (l < 7)
                gemm_launch<EPI_BIAS_BN_LRELU,false,4>(stream, tbuf, W2t, fo,
                    b2, gam, bet, xbuf, NROWS, fo, fo);
            else
                gemm_launch<EPI_BIAS,false,4>(stream, tbuf, W2t, fo,
                    b2, nullptr, nullptr, xbuf, NROWS, fo, fo);
        }
    }

    // ---- final MLP (bias-free): 128->256 lrelu ->128 lrelu ->10
    gemm_launch<EPI_LRELU,false,4>(stream, xbuf, ws + wt_off[22], 128,
        nullptr, nullptr, nullptr, big, NROWS, 256, 256);
    gemm_launch<EPI_LRELU,false,4>(stream, big, ws + wt_off[23], 256,
        nullptr, nullptr, nullptr, tbuf, NROWS, 128, 128);
    float* out10 = (float*)hhbuf;   // reuse hh buffer as [N,16] f32
    gemm_launch<EPI_NONE,true,1>(stream, tbuf, ws + wt_off[24], 128,
        nullptr, nullptr, nullptr, out10, NROWS, 10, 16);

    // ---- scatter to d_out
    {
        int n = BATCH * MAXN * 5 + BATCH * NPAIR * 5;
        scatter_out<<<dim3((n + 255) / 256), dim3(256), 0, stream>>>(out10, outp);
    }
}
